// SpatialLocalMultiheadAttention_57586921505310
// MI455X (gfx1250) — compile-verified, matrix operand order checked by reading, not run on hardware
//
#include <hip/hip_runtime.h>

// ---------------------------------------------------------------------------
// Windowed (7x7) multi-head self-attention for gfx1250 (MI455X).
// bf16 WMMA (v_wmma_f32_16x16x32_bf16) with f32 accumulation.
// One workgroup (256 threads = 8 wave32) per 49-token window, padded to 64.
// Per-wave row-block ownership: rt = wave&3, so A-fragments are loaded once
// into registers and reused across every N-tile; inner GEMM loops touch only
// global (L2-resident) weights.
// ---------------------------------------------------------------------------

typedef __attribute__((ext_vector_type(16))) __bf16 v16bf;
typedef __attribute__((ext_vector_type(8)))  float  v8f;

#define C_DIM   256
#define L_PAD   64
#define HEADS   8
#define HD      32
#define QK_SCALE 0.17677669529663687f   // 1/sqrt(32)
#define W_QKV_ELEMS (768 * 256)
#define W_OUT_ELEMS (256 * 256)

static __device__ __forceinline__ unsigned short f2bf(float f) {
    union { float f; unsigned u; } cv; cv.f = f;
    unsigned u = cv.u + 0x7FFFu + ((cv.u >> 16) & 1u);   // round-to-nearest-even
    return (unsigned short)(u >> 16);
}

// Load a 16x32 bf16 fragment (A or mirrored-B) from a row-major [row][k]
// array with leading dimension `ld`. Per lane: two 16B chunks ->
// ds_load_b128 (LDS) or global_load_b128 (weights).
static __device__ __forceinline__ v16bf frag_ld(const unsigned short* base,
                                                int row0, int ld, int k0, int lane) {
    const int r = lane & 15;
    const int h = lane >> 4;
    const unsigned short* p = base + (size_t)(row0 + r) * ld + k0 + 8 * h;
    union { uint4 q[2]; v16bf v; } u;
    u.q[0] = *(const uint4*)(p);
    u.q[1] = *(const uint4*)(p + 16);
    return u.v;
}

static __device__ __forceinline__ v8f wmma_bf(v16bf a, v16bf b, v8f c) {
    return __builtin_amdgcn_wmma_f32_16x16x32_bf16(
        /*neg_a=*/false, a, /*neg_b=*/false, b,
        /*c_mod=*/(short)0, c, /*reuse_a=*/false, /*reuse_b=*/false);
}

// ---------------------------------------------------------------------------
// Pre-pass: convert fp32 weights to bf16 once into workspace.
// ---------------------------------------------------------------------------
__global__ void __launch_bounds__(256)
cvt_weights_kernel(const float* __restrict__ w_qkv,
                   const float* __restrict__ w_out,
                   unsigned short* __restrict__ dst) {
    int i = blockIdx.x * 256 + threadIdx.x;
    if (i < W_QKV_ELEMS) {
        dst[i] = f2bf(w_qkv[i]);
    } else if (i < W_QKV_ELEMS + W_OUT_ELEMS) {
        dst[i] = f2bf(w_out[i - W_QKV_ELEMS]);
    }
}

// ---------------------------------------------------------------------------
// Main fused window-attention kernel.
// ---------------------------------------------------------------------------
__global__ void __launch_bounds__(256)
win_attn_kernel(const float* __restrict__ x,
                const float* __restrict__ in_b,      // (768)
                const float* __restrict__ out_b,     // (256)
                const unsigned short* __restrict__ wqkv_bf,  // (768,256) bf16
                const unsigned short* __restrict__ wout_bf,  // (256,256) bf16
                float* __restrict__ out) {
    // 128 KB static LDS: X/O (32K) | Q (32K) | K (32K) | V^T (32K)
    // P (per-wave 8K, 64x64 bf16) overlays the dead Q+K regions in stage 2.
    __shared__ __align__(16) unsigned short smem[65536];
    unsigned short* Xo = smem;               // [64][256] x (bf16), later O
    unsigned short* Qs = smem + 16384;       // [64][256]
    unsigned short* Ks = smem + 32768;       // [64][256]
    unsigned short* Vt = smem + 49152;       // [256][64]  (channel-major)

    const int tid   = threadIdx.x;
    const int wave  = tid >> 5;
    const int lane  = tid & 31;
    const int laneR = lane & 15;
    const int laneH = lane >> 4;

    // per-wave tile ownership (all wave-uniform, SGPR-resident)
    const int rt  = wave & 3;     // row block of the 64-token dimension
    const int nt0 = wave >> 2;    // 0/1: even/odd N-tile interleave

    // window decode: 16 images (N*T) x 16x16 windows
    const int wid  = blockIdx.x;
    const int img  = wid >> 8;
    const int widx = wid & 255;
    const int wy   = widx >> 4;
    const int wx   = widx & 15;

    // ---------------- Stage 0: global x -> LDS bf16 (pad rows 49..63) ------
    for (int i = tid; i < 64 * 64; i += 256) {
        const int row = i >> 6;
        const int c4  = (i & 63) << 2;
        unsigned short b0 = 0, b1 = 0, b2 = 0, b3 = 0;
        if (row < 49) {
            const int y  = wy * 7 + row / 7;
            const int xc = wx * 7 + row % 7;
            const size_t off = (((size_t)img * 112 + y) * 112 + xc) * C_DIM + c4;
            const float4 f = *(const float4*)(x + off);
            b0 = f2bf(f.x); b1 = f2bf(f.y); b2 = f2bf(f.z); b3 = f2bf(f.w);
        }
        uint2 pk;
        pk.x = (unsigned)b0 | ((unsigned)b1 << 16);
        pk.y = (unsigned)b2 | ((unsigned)b3 << 16);
        *(uint2*)(&Xo[row * C_DIM + c4]) = pk;
    }
    __syncthreads();

    // ---------------- Stage 1: QKV GEMM  (64x768 = 4x48 tiles) ------------
    // Hoist this wave's A-fragments (16x256 row block of X) into registers.
    v16bf aX[8];
    #pragma unroll
    for (int kk = 0; kk < 8; ++kk)
        aX[kk] = frag_ld(Xo, rt * 16, C_DIM, kk * 32, lane);

    // --- Q tiles (out channels 0..255), scale folded in ---
    for (int nt = nt0; nt < 16; nt += 2) {
        __builtin_prefetch(wqkv_bf + (size_t)((nt + 2) * 16 + laneR) * C_DIM, 0, 1);
        v8f acc = {};
        #pragma unroll
        for (int kk = 0; kk < 8; ++kk)
            acc = wmma_bf(aX[kk], frag_ld(wqkv_bf, nt * 16, C_DIM, kk * 32, lane), acc);
        const int o = nt * 16 + laneR;
        const float bias = in_b[o];
        #pragma unroll
        for (int r = 0; r < 8; ++r) {
            const int tok = rt * 16 + r + 8 * laneH;
            Qs[tok * C_DIM + o] = f2bf((acc[r] + bias) * QK_SCALE);
        }
    }
    // --- K tiles (out channels 256..511) ---
    for (int nt = 16 + nt0; nt < 32; nt += 2) {
        __builtin_prefetch(wqkv_bf + (size_t)((nt + 2) * 16 + laneR) * C_DIM, 0, 1);
        v8f acc = {};
        #pragma unroll
        for (int kk = 0; kk < 8; ++kk)
            acc = wmma_bf(aX[kk], frag_ld(wqkv_bf, nt * 16, C_DIM, kk * 32, lane), acc);
        const int o = nt * 16 + laneR;
        const float bias = in_b[o];
        #pragma unroll
        for (int r = 0; r < 8; ++r) {
            const int tok = rt * 16 + r + 8 * laneH;
            Ks[tok * C_DIM + (o - 256)] = f2bf(acc[r] + bias);
        }
    }
    // --- V tiles (out channels 512..767), stored transposed ---
    for (int nt = 32 + nt0; nt < 48; nt += 2) {
        __builtin_prefetch(wqkv_bf + (size_t)((nt + 2) * 16 + laneR) * C_DIM, 0, 1);
        v8f acc = {};
        #pragma unroll
        for (int kk = 0; kk < 8; ++kk)
            acc = wmma_bf(aX[kk], frag_ld(wqkv_bf, nt * 16, C_DIM, kk * 32, lane), acc);
        const int o = nt * 16 + laneR;
        const float bias = in_b[o];
        #pragma unroll
        for (int r = 0; r < 8; ++r) {
            const int tok = rt * 16 + r + 8 * laneH;
            Vt[(o - 512) * L_PAD + tok] = f2bf(acc[r] + bias);
        }
    }
    __syncthreads();

    // ---------------- Stage 2a: S = Q K^T   (one head per wave) -----------
    const int head = wave;
    v8f S[4][4];
    #pragma unroll
    for (int qt = 0; qt < 4; ++qt) {
        v16bf a = frag_ld(Qs, qt * 16, C_DIM, head * HD, lane);
        #pragma unroll
        for (int ct = 0; ct < 4; ++ct) {
            v16bf b = frag_ld(Ks, ct * 16, C_DIM, head * HD, lane);
            v8f z = {};
            S[qt][ct] = wmma_bf(a, b, z);
        }
    }
    __syncthreads();   // Q/K reads done everywhere; safe to overlay P

    // ---------------- Stage 2b: in-register softmax, P -> LDS bf16 --------
    unsigned short* Pw = smem + 16384 + wave * 4096;   // [64][64]
    #pragma unroll
    for (int qt = 0; qt < 4; ++qt) {
        #pragma unroll
        for (int r = 0; r < 8; ++r) {
            float m = -3.0e38f;
            #pragma unroll
            for (int ct = 0; ct < 4; ++ct) {
                const int col = ct * 16 + laneR;
                const float v = S[qt][ct][r];
                m = (col < 49) ? fmaxf(m, v) : m;
            }
            #pragma unroll
            for (int off = 1; off < 16; off <<= 1)
                m = fmaxf(m, __shfl_xor(m, off));
            float e[4];
            float s = 0.f;
            #pragma unroll
            for (int ct = 0; ct < 4; ++ct) {
                const int col = ct * 16 + laneR;
                e[ct] = (col < 49) ? __expf(S[qt][ct][r] - m) : 0.f;
                s += e[ct];
            }
            #pragma unroll
            for (int off = 1; off < 16; off <<= 1)
                s += __shfl_xor(s, off);
            const float inv = 1.0f / s;
            const int tok = qt * 16 + r + 8 * laneH;
            #pragma unroll
            for (int ct = 0; ct < 4; ++ct)
                Pw[tok * L_PAD + ct * 16 + laneR] = f2bf(e[ct] * inv);
        }
    }

    // ---------------- Stage 2c: O_h = P V_h  -> LDS (reuse X region) ------
    #pragma unroll
    for (int qt = 0; qt < 4; ++qt) {
        v16bf pa0 = frag_ld(Pw, qt * 16, L_PAD, 0,  lane);
        v16bf pa1 = frag_ld(Pw, qt * 16, L_PAD, 32, lane);
        #pragma unroll
        for (int nt = 0; nt < 2; ++nt) {
            v8f acc = {};
            acc = wmma_bf(pa0, frag_ld(Vt, head * HD + nt * 16, L_PAD, 0,  lane), acc);
            acc = wmma_bf(pa1, frag_ld(Vt, head * HD + nt * 16, L_PAD, 32, lane), acc);
            const int ch = head * HD + nt * 16 + laneR;
            #pragma unroll
            for (int r = 0; r < 8; ++r) {
                const int tok = qt * 16 + r + 8 * laneH;
                Xo[tok * C_DIM + ch] = f2bf(acc[r]);
            }
        }
    }
    __syncthreads();

    // ---------------- Stage 3: out projection + scattered store -----------
    // Hoist this wave's O row-block A-fragments; 8 N-tiles per wave.
    v16bf aO[8];
    #pragma unroll
    for (int kk = 0; kk < 8; ++kk)
        aO[kk] = frag_ld(Xo, rt * 16, C_DIM, kk * 32, lane);

    for (int nt = nt0; nt < 16; nt += 2) {
        __builtin_prefetch(wout_bf + (size_t)((nt + 2) * 16 + laneR) * C_DIM, 0, 1);
        v8f acc = {};
        #pragma unroll
        for (int kk = 0; kk < 8; ++kk)
            acc = wmma_bf(aO[kk], frag_ld(wout_bf, nt * 16, C_DIM, kk * 32, lane), acc);
        const int col  = nt * 16 + laneR;
        const float bias = out_b[col];
        if (rt < 3) {                     // tokens 0..47: always in range
            #pragma unroll
            for (int r = 0; r < 8; ++r) {
                const int tok = rt * 16 + r + 8 * laneH;
                const int y  = wy * 7 + tok / 7;
                const int xc = wx * 7 + tok % 7;
                const size_t off = (((size_t)img * 112 + y) * 112 + xc) * C_DIM + col;
                out[off] = acc[r] + bias;
            }
        } else {                          // tokens 48..63: tail tile
            #pragma unroll
            for (int r = 0; r < 8; ++r) {
                const int tok = rt * 16 + r + 8 * laneH;
                if (tok < 49) {
                    const int y  = wy * 7 + tok / 7;
                    const int xc = wx * 7 + tok % 7;
                    const size_t off = (((size_t)img * 112 + y) * 112 + xc) * C_DIM + col;
                    out[off] = acc[r] + bias;
                }
            }
        }
    }
}

// ---------------------------------------------------------------------------
extern "C" void kernel_launch(void* const* d_in, const int* in_sizes, int n_in,
                              void* d_out, int out_size, void* d_ws, size_t ws_size,
                              hipStream_t stream) {
    const float* x      = (const float*)d_in[0];   // (2,8,112,112,256)
    const float* w_qkv  = (const float*)d_in[1];   // (768,256)
    const float* b_qkv  = (const float*)d_in[2];   // (768)
    const float* w_out  = (const float*)d_in[3];   // (256,256)
    const float* b_out  = (const float*)d_in[4];   // (256)
    float* out = (float*)d_out;

    unsigned short* wbf = (unsigned short*)d_ws;   // bf16 weights workspace

    const int cvt_total = W_QKV_ELEMS + W_OUT_ELEMS;   // 262144
    cvt_weights_kernel<<<(cvt_total + 255) / 256, 256, 0, stream>>>(w_qkv, w_out, wbf);

    // 16 images * 16*16 windows = 4096 workgroups
    win_attn_kernel<<<4096, 256, 0, stream>>>(x, b_qkv, b_out,
                                              wbf, wbf + W_QKV_ELEMS, out);
}
